// PVConv_43714177139078
// MI455X (gfx1250) — matrix operand order repaired and checked by reading, not compiled, not run on hardware
//
#include <hip/hip_runtime.h>
#include <hip/hip_bf16.h>
#include <math.h>

typedef __attribute__((ext_vector_type(16))) __bf16 v16bf;
typedef __attribute__((ext_vector_type(8)))  float  v8f;
typedef int v4i __attribute__((ext_vector_type(4)));

#define BATCH 8
#define CH    64
#define RES   32
#define R3    32768      // 32^3
#define NPTS  16384

#if __has_builtin(__builtin_amdgcn_global_load_async_to_lds_b128)
#define HAS_ASYNC_LDS 1
typedef __attribute__((address_space(1))) v4i* gp_v4i;
typedef __attribute__((address_space(3))) v4i* lp_v4i;
#endif

__device__ __forceinline__ unsigned short f32_bf16(float f) {
  unsigned int u = __float_as_uint(f);
  u += 0x7FFFu + ((u >> 16) & 1u);           // round-to-nearest-even
  return (unsigned short)(u >> 16);
}
__device__ __forceinline__ int iclamp(int v, int lo, int hi) {
  return v < lo ? lo : (v > hi ? hi : v);
}

// ---------------------------------------------------------------- voxelize --
__global__ __launch_bounds__(256) void coord_stats_kernel(
    const float* __restrict__ coords, float* __restrict__ st4) {
  int b = blockIdx.x, t = threadIdx.x;
  const float* cx = coords + (size_t)b * 3 * NPTS;
  const float* cy = cx + NPTS;
  const float* cz = cy + NPTS;
  float sx = 0.f, sy = 0.f, sz = 0.f;
  for (int i = t; i < NPTS; i += 256) { sx += cx[i]; sy += cy[i]; sz += cz[i]; }
  __shared__ float sh[3][256];
  sh[0][t] = sx; sh[1][t] = sy; sh[2][t] = sz;
  __syncthreads();
  for (int o = 128; o > 0; o >>= 1) {
    if (t < o) { sh[0][t] += sh[0][t+o]; sh[1][t] += sh[1][t+o]; sh[2][t] += sh[2][t+o]; }
    __syncthreads();
  }
  float mx = sh[0][0] * (1.0f/NPTS), my = sh[1][0] * (1.0f/NPTS), mz = sh[2][0] * (1.0f/NPTS);
  __syncthreads();
  float md = 0.f;
  for (int i = t; i < NPTS; i += 256) {
    float dx = cx[i]-mx, dy = cy[i]-my, dz = cz[i]-mz;
    md = fmaxf(md, dx*dx + dy*dy + dz*dz);
  }
  sh[0][t] = md; __syncthreads();
  for (int o = 128; o > 0; o >>= 1) {
    if (t < o) sh[0][t] = fmaxf(sh[0][t], sh[0][t+o]);
    __syncthreads();
  }
  if (t == 0) {
    st4[b*4+0] = mx; st4[b*4+1] = my; st4[b*4+2] = mz;
    st4[b*4+3] = sqrtf(sh[0][0]) * 2.0f;
  }
}

__global__ __launch_bounds__(256) void compute_nc_kernel(
    const float* __restrict__ coords, const float* __restrict__ st4,
    float* __restrict__ nc, int* __restrict__ vox) {
  int t = blockIdx.x * 256 + threadIdx.x;
  if (t >= BATCH * NPTS) return;
  int b = t >> 14, n = t & (NPTS - 1);
  float inv = 1.0f / st4[b*4+3];
  int vi[3];
#pragma unroll
  for (int d = 0; d < 3; ++d) {
    float v = coords[((size_t)(b*3 + d)) * NPTS + n];
    float x = (v - st4[b*4+d]) * inv + 0.5f;
    x = fminf(fmaxf(x * (float)RES, 0.0f), (float)(RES - 1));
    nc[((size_t)(b*3 + d)) * NPTS + n] = x;
    vi[d] = (int)rintf(x);
  }
  vox[t] = (vi[0] * RES + vi[1]) * RES + vi[2];
}

__global__ __launch_bounds__(256) void scatter_kernel(
    const float* __restrict__ feats, const int* __restrict__ vox,
    float* __restrict__ sums, float* __restrict__ cnt) {
  int t = blockIdx.x * 256 + threadIdx.x;
  if (t >= BATCH * NPTS) return;
  int b = t >> 14, n = t & (NPTS - 1);
  int idx = vox[t];
  atomicAdd(&cnt[b * R3 + idx], 1.0f);
  float* srow = sums + ((size_t)(b * R3 + idx)) * CH;        // (B, XYZ, C)
  const float* f = feats + (size_t)b * CH * NPTS + n;
  for (int c = 0; c < CH; ++c)
    atomicAdd(&srow[c], f[(size_t)c * NPTS]);
}

__global__ __launch_bounds__(256) void finalize_grid_kernel(
    const float* __restrict__ sums, const float* __restrict__ cnt,
    unsigned short* __restrict__ gbf) {
  size_t t = (size_t)blockIdx.x * 256 + threadIdx.x;         // B*R3*CH
  int b = (int)(t >> 21);
  int p = (int)((t >> 6) & (R3 - 1));
  float cv = cnt[b * R3 + p];
  gbf[t] = f32_bf16(sums[t] / fmaxf(cv, 1.0f));              // (B, XYZ, C) bf16
}

// -------------------------------------------------------- weight packing --
// B-fragment order for V_WMMA_F32_16X16X32_BF16:
//   lane holds col N = lane&15; element h maps to K = (lane>>4)*8 + (h&7) + (h>>3)*16
__global__ __launch_bounds__(256) void pack_conv_w_kernel(
    const float* __restrict__ w, unsigned short* __restrict__ wf) {
  int t = blockIdx.x * 256 + threadIdx.x;
  if (t >= 54 * 4 * 32 * 16) return;
  int h = t & 15, lane = (t >> 4) & 31, n = (t >> 9) & 3, c = t >> 11;
  int g = lane >> 4, m = lane & 15;
  int cout = n * 16 + m;
  int cin = ((c & 1) << 5) + g * 8 + (h & 7) + ((h >> 3) << 4);
  int tap = c >> 1;
  int tx = tap / 9, ty = (tap / 3) % 3, tz = tap % 3;
  wf[t] = f32_bf16(w[(size_t)(cout * 64 + cin) * 27 + tx * 9 + ty * 3 + tz]);
}

__global__ __launch_bounds__(256) void pack_pf_w_kernel(
    const float* __restrict__ w, unsigned short* __restrict__ wf) {
  int t = blockIdx.x * 256 + threadIdx.x;
  if (t >= 2 * 4 * 32 * 16) return;
  int h = t & 15, lane = (t >> 4) & 31, n = (t >> 9) & 3, c = t >> 11;
  int g = lane >> 4, m = lane & 15;
  int cout = n * 16 + m;
  int cin = (c << 5) + g * 8 + (h & 7) + ((h >> 3) << 4);
  wf[t] = f32_bf16(w[cout * 64 + cin]);
}

// ------------------------------------------------- conv3d as implicit GEMM --
// block = 1 wave (32 thr) = one (b,x,y) row of 32 voxels = two 16-row z-tiles.
// K = 27 taps x 64 Cin = 54 chunks of 32.
// Per chunk: 2 A-frags (4 ds_load_b128), 4 B-frags (8 global_load_b128, each
// reused for both z-tiles) -> 8 WMMA. Halo tile staged via async-to-LDS DMA.
__global__ __launch_bounds__(32) void conv3d_wmma_kernel(
    const unsigned short* __restrict__ gin,   // (B, X,Y,Z, C) bf16
    const unsigned short* __restrict__ wf,    // [54][4][32][16] bf16
    const float* __restrict__ bias,
    float* __restrict__ out) {                // (B, C, XYZ) f32
  __shared__ uint4 tile4[9 * 34 * 8];         // [nxy][z+1][c] : 3x3 halo, z=-1..32
  const int bi = blockIdx.x;
  const int b = bi >> 10, x = (bi >> 5) & 31, y = bi & 31;
  const int tid = threadIdx.x;

  const size_t gbase = (size_t)b * R3 * CH;
  for (int u = tid; u < 9 * 34 * 8; u += 32) {
    int nxy = u / (34 * 8);
    int rem = u - nxy * (34 * 8);
    int zi = rem >> 3, q = rem & 7;
    int xx = x + nxy / 3 - 1, yy = y + nxy % 3 - 1, zz = zi - 1;
    bool inb = (xx >= 0 && xx < 32 && yy >= 0 && yy < 32 && zz >= 0 && zz < 32);
#ifdef HAS_ASYNC_LDS
    if (inb) {
      const unsigned short* src =
          gin + gbase + (size_t)((xx * 32 + yy) * 32 + zz) * CH + q * 8;
      __builtin_amdgcn_global_load_async_to_lds_b128(
          (gp_v4i)src, (lp_v4i)&tile4[u], 0, 0);
    } else {
      tile4[u] = make_uint4(0u, 0u, 0u, 0u);
    }
#else
    uint4 v = make_uint4(0u, 0u, 0u, 0u);
    if (inb)
      v = *(const uint4*)(gin + gbase + (size_t)((xx * 32 + yy) * 32 + zz) * CH + q * 8);
    tile4[u] = v;
#endif
  }
#ifdef HAS_ASYNC_LDS
#if __has_builtin(__builtin_amdgcn_s_wait_asynccnt)
  __builtin_amdgcn_s_wait_asynccnt(0);
#else
  asm volatile("s_wait_asynccnt 0x0" ::: "memory");
#endif
#endif
  __syncthreads();                            // single-wave WG: barrier is free

  const unsigned short* tile = (const unsigned short*)tile4;
  const int lane = tid & 31;
  const int m = lane & 15, g = lane >> 4;

  v8f acc[8] = {};                            // [ztile 2][coutChunk 4]
  for (int c = 0; c < 54; ++c) {
    int tap = c >> 1;
    int tz = tap % 3, ty = (tap / 3) % 3, tx = tap / 9;
    int cinb = ((c & 1) << 5) + (g << 3);
    const unsigned short* ap0 = tile + ((tx * 3 + ty) * 34 + m + tz) * 64 + cinb;
    union { uint4 u[2]; v16bf v; } A0, A1;
    A0.u[0] = *(const uint4*)ap0;
    A0.u[1] = *(const uint4*)(ap0 + 16);
    A1.u[0] = *(const uint4*)(ap0 + 16 * 64);       // z-tile 1: zi += 16
    A1.u[1] = *(const uint4*)(ap0 + 16 * 64 + 16);
#pragma unroll
    for (int n = 0; n < 4; ++n) {
      union { uint4 u[2]; v16bf v; } Bf;
      const uint4* w4 = (const uint4*)(wf + (((c * 4 + n) * 32 + lane) << 4));
      Bf.u[0] = w4[0]; Bf.u[1] = w4[1];
      acc[n]     = __builtin_amdgcn_wmma_f32_16x16x32_bf16(
          false, A0.v, false, Bf.v, (short)0, acc[n], false, false);
      acc[4 + n] = __builtin_amdgcn_wmma_f32_16x16x32_bf16(
          false, A1.v, false, Bf.v, (short)0, acc[4 + n], false, false);
    }
  }

#pragma unroll
  for (int zt = 0; zt < 2; ++zt) {
    const int pbase = x * 1024 + y * 32 + zt * 16 + g * 8;  // lane owns M=g*8+v
#pragma unroll
    for (int n = 0; n < 4; ++n) {
      const v8f a = acc[zt * 4 + n];
      int cout = n * 16 + m;
      float bs = bias[cout];
      float* op = out + ((size_t)(b * CH + cout)) * R3 + pbase;
      *(float4*)op     = make_float4(a[0]+bs, a[1]+bs, a[2]+bs, a[3]+bs);
      *(float4*)(op+4) = make_float4(a[4]+bs, a[5]+bs, a[6]+bs, a[7]+bs);
    }
  }
}

// --------------------------------------------------- point branch 1x1 GEMM --
__global__ __launch_bounds__(256) void pf_wmma_kernel(
    const float* __restrict__ feats,          // (B, 64, N) f32
    const unsigned short* __restrict__ wf,    // [2][4][32][16] bf16
    const float* __restrict__ bias,
    float* __restrict__ pout) {               // (B, 64, N) f32
  __shared__ uint4 pt4[8 * 16 * 8];           // [tile(8)][m(16)][c(64)] bf16
  unsigned short* pt = (unsigned short*)pt4;
  int b = blockIdx.x >> 7;
  int n0 = (blockIdx.x & 127) << 7;           // 128 points per block
  const float* fb = feats + (size_t)b * CH * NPTS;
  for (int i = threadIdx.x; i < 64 * 128; i += 256) {
    int c = i >> 7, p = i & 127;
    pt[(p << 6) + c] = f32_bf16(fb[(size_t)c * NPTS + n0 + p]);
  }
  __syncthreads();

  int wave = threadIdx.x >> 5, lane = threadIdx.x & 31;
  int m = lane & 15, g = lane >> 4;
  v8f acc[4] = {};
#pragma unroll
  for (int c = 0; c < 2; ++c) {
    const unsigned short* ap = pt + (((wave << 4) + m) << 6) + (c << 5) + (g << 3);
    union { uint4 u[2]; v16bf v; } A;
    A.u[0] = *(const uint4*)ap;
    A.u[1] = *(const uint4*)(ap + 16);
#pragma unroll
    for (int n = 0; n < 4; ++n) {
      union { uint4 u[2]; v16bf v; } Bf;
      const uint4* w4 = (const uint4*)(wf + (((c * 4 + n) * 32 + lane) << 4));
      Bf.u[0] = w4[0]; Bf.u[1] = w4[1];
      acc[n] = __builtin_amdgcn_wmma_f32_16x16x32_bf16(
          false, A.v, false, Bf.v, (short)0, acc[n], false, false);
    }
  }
  int pbase = n0 + (wave << 4) + (g << 3);
#pragma unroll
  for (int n = 0; n < 4; ++n) {
    int cout = n * 16 + m;
    float bs = bias[cout];
    float* op = pout + ((size_t)(b * CH + cout)) * NPTS + pbase;
    *(float4*)op     = make_float4(acc[n][0]+bs, acc[n][1]+bs, acc[n][2]+bs, acc[n][3]+bs);
    *(float4*)(op+4) = make_float4(acc[n][4]+bs, acc[n][5]+bs, acc[n][6]+bs, acc[n][7]+bs);
  }
}

// ------------------------------------------------------------- group norm --
__global__ __launch_bounds__(256) void gn_stats_kernel(
    const float* __restrict__ x, float* __restrict__ st, int S) {
  int b = blockIdx.x >> 3, g = blockIdx.x & 7;
  const float4* p = (const float4*)(x + (size_t)(b * 64 + g * 8) * S);
  int n4 = (8 * S) >> 2;
  float s = 0.f, ss = 0.f;
  for (int i = threadIdx.x; i < n4; i += 256) {
    float4 v = p[i];
    s  += v.x + v.y + v.z + v.w;
    ss += v.x*v.x + v.y*v.y + v.z*v.z + v.w*v.w;
  }
  __shared__ float sh[256], sh2[256];
  sh[threadIdx.x] = s; sh2[threadIdx.x] = ss;
  __syncthreads();
  for (int o = 128; o > 0; o >>= 1) {
    if (threadIdx.x < o) { sh[threadIdx.x] += sh[threadIdx.x+o]; sh2[threadIdx.x] += sh2[threadIdx.x+o]; }
    __syncthreads();
  }
  if (threadIdx.x == 0) {
    float cntv = 8.0f * (float)S;
    float mu = sh[0] / cntv;
    float var = sh2[0] / cntv - mu * mu;
    st[blockIdx.x * 2]     = mu;
    st[blockIdx.x * 2 + 1] = rsqrtf(var + 1e-5f);
  }
}

// GN + swish, (B,C,XYZ) f32 -> (B,XYZ,C) bf16 (input layout for conv2)
__global__ __launch_bounds__(256) void gn_swish_bf16_kernel(
    const float* __restrict__ x, const float* __restrict__ st,
    const float* __restrict__ w, const float* __restrict__ bb,
    unsigned short* __restrict__ out) {
  size_t t = (size_t)blockIdx.x * 256 + threadIdx.x;         // B*R3*CH
  int b = (int)(t >> 21);
  int rem = (int)(t & ((1u << 21) - 1));
  int p = rem >> 6, c = rem & 63;
  float v = x[((size_t)(b * CH + c) << 15) + p];
  int si = b * 8 + (c >> 3);
  float yv = (v - st[si*2]) * st[si*2+1] * w[c] + bb[c];
  out[t] = f32_bf16(yv / (1.0f + __expf(-yv)));
}

// GN in place (no swish), layout (B,C,XYZ)
__global__ __launch_bounds__(256) void gn_apply_kernel(
    float* __restrict__ x, const float* __restrict__ st,
    const float* __restrict__ w, const float* __restrict__ bb) {
  size_t t = (size_t)blockIdx.x * 256 + threadIdx.x;
  int b = (int)(t >> 21);
  int c = (int)((t >> 15) & 63);
  int si = b * 8 + (c >> 3);
  x[t] = (x[t] - st[si*2]) * st[si*2+1] * w[c] + bb[c];
}

// ------------------------------------ devoxelize + point-branch GN/swish ---
__global__ __launch_bounds__(256) void final_kernel(
    const float* __restrict__ g,   // (B, C, XYZ) f32
    const float* __restrict__ p,   // (B, C, N)   f32
    const float* __restrict__ nc,  // (B, 3, N)
    const float* __restrict__ pst,
    const float* __restrict__ pgw, const float* __restrict__ pgb,
    float* __restrict__ out) {     // (B, C, N)
  size_t t = (size_t)blockIdx.x * 256 + threadIdx.x;         // B*CH*NPTS
  int b = (int)(t >> 20);
  int c = (int)((t >> 14) & 63);
  int n = (int)(t & (NPTS - 1));
  const float* ncb = nc + (size_t)b * 3 * NPTS + n;
  float xf = ncb[0], yf = ncb[NPTS], zf = ncb[2 * NPTS];
  float fx = floorf(xf), fy = floorf(yf), fz = floorf(zf);
  int x0 = iclamp((int)fx, 0, 31), y0 = iclamp((int)fy, 0, 31), z0 = iclamp((int)fz, 0, 31);
  int x1 = iclamp(x0 + 1, 0, 31), y1 = iclamp(y0 + 1, 0, 31), z1 = iclamp(z0 + 1, 0, 31);
  float wx1 = xf - fx, wy1 = yf - fy, wz1 = zf - fz;
  float wx0 = 1.f - wx1, wy0 = 1.f - wy1, wz0 = 1.f - wz1;
  const float* gp = g + ((size_t)(b * CH + c) << 15);
  float a = 0.f;
  a += gp[(x0*32 + y0)*32 + z0] * wx0 * wy0 * wz0;
  a += gp[(x0*32 + y0)*32 + z1] * wx0 * wy0 * wz1;
  a += gp[(x0*32 + y1)*32 + z0] * wx0 * wy1 * wz0;
  a += gp[(x0*32 + y1)*32 + z1] * wx0 * wy1 * wz1;
  a += gp[(x1*32 + y0)*32 + z0] * wx1 * wy0 * wz0;
  a += gp[(x1*32 + y0)*32 + z1] * wx1 * wy0 * wz1;
  a += gp[(x1*32 + y1)*32 + z0] * wx1 * wy1 * wz0;
  a += gp[(x1*32 + y1)*32 + z1] * wx1 * wy1 * wz1;
  float pv = p[((size_t)(b * CH + c) << 14) + n];
  int si = b * 8 + (c >> 3);
  float yv = (pv - pst[si*2]) * pst[si*2+1] * pgw[c] + pgb[c];
  a += yv / (1.0f + __expf(-yv));
  out[t] = a;
}

// ------------------------------------------------------------------ host ---
extern "C" void kernel_launch(void* const* d_in, const int* in_sizes, int n_in,
                              void* d_out, int out_size, void* d_ws, size_t ws_size,
                              hipStream_t stream) {
  (void)in_sizes; (void)n_in; (void)out_size; (void)ws_size;
  const float* features = (const float*)d_in[0];
  const float* coords   = (const float*)d_in[1];
  const float* conv1_w  = (const float*)d_in[2];
  const float* conv1_b  = (const float*)d_in[3];
  const float* gn1_w    = (const float*)d_in[4];
  const float* gn1_b    = (const float*)d_in[5];
  const float* conv2_w  = (const float*)d_in[6];
  const float* conv2_b  = (const float*)d_in[7];
  const float* gn2_w    = (const float*)d_in[8];
  const float* gn2_b    = (const float*)d_in[9];
  const float* pf_w     = (const float*)d_in[10];
  const float* pf_b     = (const float*)d_in[11];
  const float* pf_gn_w  = (const float*)d_in[12];
  const float* pf_gn_b  = (const float*)d_in[13];
  float* out = (float*)d_out;

  char* ws = (char*)d_ws;
  const size_t szA   = (size_t)BATCH * CH * R3 * 4;   // 64 MB: sums -> conv out -> g
  const size_t szCnt = (size_t)BATCH * R3 * 4;        // 1 MB
  const size_t szG1  = (size_t)BATCH * R3 * CH * 2;   // 32 MB bf16 grid
  const size_t szNc  = (size_t)BATCH * 3 * NPTS * 4;
  const size_t szVox = (size_t)BATCH * NPTS * 4;
  const size_t szP   = (size_t)BATCH * CH * NPTS * 4; // 32 MB
  const size_t szWc  = (size_t)54 * 4 * 32 * 16 * 2;  // 216 KB
  const size_t szWp  = (size_t)2 * 4 * 32 * 16 * 2;

  size_t o = 0;
  float*          bufA  = (float*)(ws + o);          o += szA;
  float*          cnt   = (float*)(ws + o);          o += szCnt;
  unsigned short* g1    = (unsigned short*)(ws + o); o += szG1;
  float*          ncBuf = (float*)(ws + o);          o += szNc;
  int*            vox   = (int*)(ws + o);            o += szVox;
  float*          pBuf  = (float*)(ws + o);          o += szP;
  unsigned short* wf1   = (unsigned short*)(ws + o); o += szWc;
  unsigned short* wf2   = (unsigned short*)(ws + o); o += szWc;
  unsigned short* wfp   = (unsigned short*)(ws + o); o += szWp;
  float*          st4   = (float*)(ws + o);          o += 256;
  float*          gnS1  = (float*)(ws + o);          o += 512;
  float*          gnS2  = (float*)(ws + o);          o += 512;
  float*          gnSP  = (float*)(ws + o);          o += 512;

  // zero scatter accumulators (sums + cnt are adjacent)
  (void)hipMemsetAsync(bufA, 0, szA + szCnt, stream);

  // weight packing (cheap, deterministic each call)
  pack_conv_w_kernel<<<432, 256, 0, stream>>>(conv1_w, wf1);
  pack_conv_w_kernel<<<432, 256, 0, stream>>>(conv2_w, wf2);
  pack_pf_w_kernel<<<16, 256, 0, stream>>>(pf_w, wfp);

  // voxelize
  coord_stats_kernel<<<BATCH, 256, 0, stream>>>(coords, st4);
  compute_nc_kernel<<<(BATCH * NPTS) / 256, 256, 0, stream>>>(coords, st4, ncBuf, vox);
  scatter_kernel<<<(BATCH * NPTS) / 256, 256, 0, stream>>>(features, vox, bufA, cnt);
  finalize_grid_kernel<<<(BATCH * R3 * CH) / 256, 256, 0, stream>>>(bufA, cnt, g1);

  // conv1 + GN + swish
  conv3d_wmma_kernel<<<BATCH * 1024, 32, 0, stream>>>(g1, wf1, conv1_b, bufA);
  gn_stats_kernel<<<BATCH * 8, 256, 0, stream>>>(bufA, gnS1, R3);
  gn_swish_bf16_kernel<<<(BATCH * R3 * CH) / 256, 256, 0, stream>>>(bufA, gnS1, gn1_w, gn1_b, g1);

  // conv2 + GN (no swish), in-place -> g
  conv3d_wmma_kernel<<<BATCH * 1024, 32, 0, stream>>>(g1, wf2, conv2_b, bufA);
  gn_stats_kernel<<<BATCH * 8, 256, 0, stream>>>(bufA, gnS2, R3);
  gn_apply_kernel<<<(BATCH * R3 * CH) / 256, 256, 0, stream>>>(bufA, gnS2, gn2_w, gn2_b);

  // point branch GEMM + its GN stats
  pf_wmma_kernel<<<BATCH * 128, 256, 0, stream>>>(features, wfp, pf_b, pBuf);
  gn_stats_kernel<<<BATCH * 8, 256, 0, stream>>>(pBuf, gnSP, NPTS);

  // trilinear devoxelize + point GN/swish + sum
  final_kernel<<<(BATCH * CH * NPTS) / 256, 256, 0, stream>>>(
      bufA, pBuf, ncBuf, gnSP, pf_gn_w, pf_gn_b, out);
}
